// RWKV7Layer_9079560864065
// MI455X (gfx1250) — matrix operand order repaired and checked by reading, not compiled, not run on hardware
//
#include <hip/hip_runtime.h>
#include <math.h>

typedef __bf16 bf16_t;
typedef __bf16 v16bf __attribute__((ext_vector_type(16)));
typedef float  v8f   __attribute__((ext_vector_type(8)));
typedef unsigned int u32x4 __attribute__((ext_vector_type(4)));
typedef int v4i_vs __attribute__((vector_size(16)));   // matches builtin param type

#define AS1 __attribute__((address_space(1)))
#define AS3 __attribute__((address_space(3)))

#if __has_builtin(__builtin_amdgcn_global_load_async_to_lds_b128)
#define RWKV_ASYNC_LDS 1
#endif

constexpr int kB = 8, kT = 2048, kC = 512, kH = 4, kK = 512, kV = 1024;
constexpr int kDK = 128, kDV = 256;
constexpr int kM = kB * kT;   // 16384 tokens

// ---------------- weight transpose + f32 -> bf16 ----------------
__global__ __launch_bounds__(256) void tconv_kernel(const float* __restrict__ src,
                                                    bf16_t* __restrict__ dst,
                                                    int R, int Ccols) {
  int idx = blockIdx.x * 256 + threadIdx.x;
  if (idx >= R * Ccols) return;
  int r = idx / Ccols, c = idx % Ccols;
  dst[(size_t)c * R + r] = (bf16_t)src[idx];   // dst is [Ccols][R] = [N][K]
}

// ---------------- LayerNorm over C=512 (one block per token) ----------------
__global__ __launch_bounds__(256) void ln_kernel(const float* __restrict__ x,
                                                 const float* __restrict__ w,
                                                 const float* __restrict__ b,
                                                 float* __restrict__ nout) {
  __shared__ float red[256];
  int m = blockIdx.x, tid = threadIdx.x;
  size_t base = (size_t)m * kC;
  float v0 = x[base + tid], v1 = x[base + tid + 256];
  red[tid] = v0 + v1;
  __syncthreads();
  for (int s = 128; s > 0; s >>= 1) { if (tid < s) red[tid] += red[tid + s]; __syncthreads(); }
  float mu = red[0] * (1.0f / kC);
  __syncthreads();
  float d0 = v0 - mu, d1 = v1 - mu;
  red[tid] = d0 * d0 + d1 * d1;
  __syncthreads();
  for (int s = 128; s > 0; s >>= 1) { if (tid < s) red[tid] += red[tid + s]; __syncthreads(); }
  float rstd = rsqrtf(red[0] * (1.0f / kC) + 1e-5f);
  nout[base + tid]       = d0 * rstd * w[tid] + b[tid];
  nout[base + tid + 256] = d1 * rstd * w[tid + 256] + b[tid + 256];
}

// ---------------- token-shift mix -> 6 bf16 activation streams --------------
__global__ __launch_bounds__(256) void mix_kernel(const float* __restrict__ n,
    const float* __restrict__ mr, const float* __restrict__ mw,
    const float* __restrict__ mk, const float* __restrict__ mv,
    const float* __restrict__ ma, const float* __restrict__ mg,
    bf16_t* __restrict__ xr, bf16_t* __restrict__ xw, bf16_t* __restrict__ xk,
    bf16_t* __restrict__ xv, bf16_t* __restrict__ xa, bf16_t* __restrict__ xg) {
  int m = blockIdx.x, tid = threadIdx.x;
  int t = m % kT;
  size_t base = (size_t)m * kC;
#pragma unroll
  for (int j = 0; j < 2; ++j) {
    int c = tid + j * 256;
    float nc = n[base + c];
    float np = (t == 0) ? 0.0f : n[base - kC + c];
    float dx = np - nc;
    xr[base + c] = (bf16_t)(nc + dx * mr[c]);
    xw[base + c] = (bf16_t)(nc + dx * mw[c]);
    xk[base + c] = (bf16_t)(nc + dx * mk[c]);
    xv[base + c] = (bf16_t)(nc + dx * mv[c]);
    xa[base + c] = (bf16_t)(nc + dx * ma[c]);
    xg[base + c] = (bf16_t)(nc + dx * mg[c]);
  }
}

// ---------------- WMMA bf16 GEMM: C[M,N] = A[M,K] * Bt[N,K]^T (+residual) ---
// 256 threads = 8 waves (4x2). Block tile BMxBN, wave tile (BM/4)x(BN/2),
// double-buffered LDS; async global->LDS when the toolchain exposes it.
template <int BM, int BN>
__global__ __launch_bounds__(256) void gemm_bf16_kernel(const bf16_t* __restrict__ A,
    const bf16_t* __restrict__ Bt, float* __restrict__ Cout,
    const float* __restrict__ residual, int N, int Kd) {
  constexpr int WM = BM / 4, WN = BN / 2;
  constexpr int NA = WM / 16, NB = WN / 16;
  constexpr int AV = BM / 64, BV = BN / 64;
  __shared__ __align__(16) bf16_t Alds[2][BM][40];  // row pad -> 16B-aligned rows
  __shared__ __align__(16) bf16_t Blds[2][BN][40];
  const int tid = threadIdx.x;
  const int m0 = blockIdx.y * BM, n0 = blockIdx.x * BN;
  const int lrow = tid >> 2, lcol = (tid & 3) * 8;   // tile loader mapping
  const int lane = tid & 31, w = tid >> 5;
  const int wm = w & 3, wn = w >> 2;
  const int lhi = lane >> 4, llo = lane & 15;
  union AF { v16bf v; u32x4 q[2]; };
  union CF { v8f v; float f[8]; };
  CF acc[NA][NB];
#pragma unroll
  for (int a = 0; a < NA; ++a)
#pragma unroll
    for (int b = 0; b < NB; ++b)
#pragma unroll
      for (int i = 0; i < 8; ++i) acc[a][b].f[i] = 0.0f;

  const bf16_t* ag = A  + (size_t)(m0 + lrow) * Kd + lcol;
  const bf16_t* bg = Bt + (size_t)(n0 + lrow) * Kd + lcol;

  auto compute = [&](int buf) {
    AF af[NA];
#pragma unroll
    for (int na = 0; na < NA; ++na) {
      // A frag 16x32: halves 0..7 = K kb..kb+7, 8..15 = K kb+16..kb+23 (kb=lhi*8)
      const bf16_t* ap = &Alds[buf][wm * WM + na * 16 + llo][lhi * 8];
      af[na].q[0] = *(const u32x4*)ap;
      af[na].q[1] = *(const u32x4*)(ap + 16);
    }
#pragma unroll
    for (int nb = 0; nb < NB; ++nb) {
      // B frag 32x16: halves 0..7 = K kb..kb+7, 8..15 = K kb+16..kb+23 (kb=lhi*8)
      AF bf;
      const bf16_t* bp = &Blds[buf][wn * WN + nb * 16 + llo][lhi * 8];
      bf.q[0] = *(const u32x4*)bp;
      bf.q[1] = *(const u32x4*)(bp + 16);
#pragma unroll
      for (int na = 0; na < NA; ++na)
        acc[na][nb].v = __builtin_amdgcn_wmma_f32_16x16x32_bf16(
            false, af[na].v, false, bf.v, (short)0, acc[na][nb].v, false, false);
    }
  };

  const int nsteps = Kd >> 5;

#ifdef RWKV_ASYNC_LDS
  auto issue = [&](int buf, int k0) {
#pragma unroll
    for (int i = 0; i < AV; ++i)
      __builtin_amdgcn_global_load_async_to_lds_b128(
          (AS1 v4i_vs*)(unsigned long long)(const void*)(ag + (size_t)i * 64 * Kd + k0),
          (AS3 v4i_vs*)&Alds[buf][i * 64 + lrow][lcol], 0, 0);
#pragma unroll
    for (int i = 0; i < BV; ++i)
      __builtin_amdgcn_global_load_async_to_lds_b128(
          (AS1 v4i_vs*)(unsigned long long)(const void*)(bg + (size_t)i * 64 * Kd + k0),
          (AS3 v4i_vs*)&Blds[buf][i * 64 + lrow][lcol], 0, 0);
  };
  issue(0, 0);
  asm volatile("s_wait_asynccnt 0x0" ::: "memory");
  __syncthreads();
  for (int s = 0; s < nsteps; ++s) {
    const int cur = s & 1;
    if (s + 1 < nsteps) issue(cur ^ 1, (s + 1) << 5);
    compute(cur);
    asm volatile("s_wait_asynccnt 0x0" ::: "memory");
    __syncthreads();
  }
#else
  u32x4 areg[AV], breg[BV];
  auto fetch = [&](int k0) {
#pragma unroll
    for (int i = 0; i < AV; ++i) areg[i] = *(const u32x4*)(ag + (size_t)i * 64 * Kd + k0);
#pragma unroll
    for (int i = 0; i < BV; ++i) breg[i] = *(const u32x4*)(bg + (size_t)i * 64 * Kd + k0);
  };
  auto commit = [&](int buf) {
#pragma unroll
    for (int i = 0; i < AV; ++i) *(u32x4*)&Alds[buf][i * 64 + lrow][lcol] = areg[i];
#pragma unroll
    for (int i = 0; i < BV; ++i) *(u32x4*)&Blds[buf][i * 64 + lrow][lcol] = breg[i];
  };
  fetch(0);
  commit(0);
  __syncthreads();
  for (int s = 0; s < nsteps; ++s) {
    const int cur = s & 1;
    if (s + 1 < nsteps) {
      fetch((s + 1) << 5);
      if (s + 2 < nsteps) {                // gfx1250 global_prefetch_b8
        __builtin_prefetch(ag + ((size_t)(s + 2) << 5), 0, 3);
        __builtin_prefetch(bg + ((size_t)(s + 2) << 5), 0, 3);
      }
    }
    compute(cur);
    if (s + 1 < nsteps) commit(cur ^ 1);
    __syncthreads();
  }
#endif

  // D layout: VGPR i -> M = i + 8*lhi ; lane llo -> N column within 16-wide tile
#pragma unroll
  for (int na = 0; na < NA; ++na) {
#pragma unroll
    for (int nb = 0; nb < NB; ++nb) {
      const int mrow = m0 + wm * WM + na * 16 + lhi * 8;
      const int ncol = n0 + wn * WN + nb * 16 + llo;
#pragma unroll
      for (int i = 0; i < 8; ++i) {
        size_t idx = (size_t)(mrow + i) * N + ncol;
        float r0 = acc[na][nb].f[i];
        if (residual) r0 += residual[idx];
        Cout[idx] = r0;
      }
    }
  }
}

// ---------------- elementwise: f32 -> activation -> bf16 ----------------
__global__ __launch_bounds__(256) void act_bf16_kernel(const float* __restrict__ src,
                                                       bf16_t* __restrict__ dst,
                                                       int count, int mode) {
  int i = blockIdx.x * 256 + threadIdx.x;
  if (i >= count) return;
  float v = src[i];
  if (mode == 0) v = tanhf(v);
  else if (mode == 1) v = 1.0f / (1.0f + expf(-v));
  dst[i] = (bf16_t)v;
}

// d = exp(-exp(-softplus(-(w0+z)) - 0.5))  (in place)
__global__ __launch_bounds__(256) void decay_kernel(const float* src,
                                                    const float* __restrict__ w0,
                                                    float* dst, int count) {
  int i = blockIdx.x * 256 + threadIdx.x;
  if (i >= count) return;
  float y = w0[i % kK] + src[i];
  float wv = -log1pf(expf(-y)) - 0.5f;
  dst[i] = expf(-expf(wv));
}

// a = sigmoid(a0 + z)  (in place)
__global__ __launch_bounds__(256) void asig_kernel(const float* src,
                                                   const float* __restrict__ a0,
                                                   float* dst, int count) {
  int i = blockIdx.x * 256 + threadIdx.x;
  if (i >= count) return;
  dst[i] = 1.0f / (1.0f + expf(-(a0[i % kK] + src[i])));
}

// kk = l2norm(k*k_k) per head; k <- k*(1+(a-1)*k_a)
__global__ __launch_bounds__(128) void kprep_kernel(float* kbuf,
    const float* __restrict__ abuf, float* __restrict__ kkbuf,
    const float* __restrict__ k_k, const float* __restrict__ k_a) {
  __shared__ float red[128];
  int bh = blockIdx.x;
  int m = bh / kH, h = bh % kH, i = threadIdx.x;
  size_t base = (size_t)m * kK + h * kDK + i;
  int hk = h * kDK + i;
  float kv = kbuf[base];
  float kkraw = kv * k_k[hk];
  red[i] = kkraw * kkraw;
  __syncthreads();
  for (int s = 64; s > 0; s >>= 1) { if (i < s) red[i] += red[i + s]; __syncthreads(); }
  float inv = 1.0f / fmaxf(sqrtf(red[0]), 1e-12f);
  kkbuf[base] = kkraw * inv;
  float av = abuf[base];
  kbuf[base] = kv * (1.0f + (av - 1.0f) * k_a[hk]);
}

// ---------------- sequential RWKV-7 scan: one block per (b,h) ----------------
// Each lane owns one state row S[v, 0..127] in registers (256 lanes = DV rows).
__global__ __launch_bounds__(256) void scan_kernel(const float* __restrict__ rbuf,
    const float* __restrict__ kbuf, const float* __restrict__ kkbuf,
    const float* __restrict__ dbuf, const float* __restrict__ abuf,
    const float* __restrict__ vbuf, float* __restrict__ obuf) {
  __shared__ float rr[kDK], kks[kDK], ks[kDK], ds[kDK], as2[kDK], vs[kDV];
  int bh = blockIdx.x;
  int b = bh / kH, h = bh % kH;
  int tid = threadIdx.x;
  float S[kDK];
#pragma unroll
  for (int k = 0; k < kDK; ++k) S[k] = 0.0f;
  size_t mk = (size_t)b * kT * kK + (size_t)h * kDK;
  size_t mv = (size_t)b * kT * kV + (size_t)h * kDV;
  for (int t = 0; t < kT; ++t, mk += kK, mv += kV) {
    int q = tid & 127;
    if (tid < 128) {
      rr[q]  = rbuf[mk + q];
      kks[q] = kkbuf[mk + q];
      ks[q]  = kbuf[mk + q];
    } else {
      ds[q]  = dbuf[mk + q];
      as2[q] = abuf[mk + q];
    }
    vs[tid] = vbuf[mv + tid];
    if (t + 1 < kT) {
      __builtin_prefetch(rbuf + mk + kK + q, 0, 3);
      __builtin_prefetch(vbuf + mv + kV + tid, 0, 3);
    }
    __syncthreads();
    float sa = 0.0f;
#pragma unroll
    for (int k = 0; k < kDK; ++k) sa += S[k] * kks[k];   // S . kk  (per row v)
    float vt = vs[tid];
    float ov = 0.0f;
#pragma unroll
    for (int k = 0; k < kDK; ++k) {
      float s2 = S[k] * ds[k] - sa * (kks[k] * as2[k]) + vt * ks[k];
      S[k] = s2;
      ov += s2 * rr[k];                                   // S_new . r
    }
    obuf[mv + tid] = ov;
    __syncthreads();
  }
}

// ------------- per-head GroupNorm + bonus + gate -> bf16 stream -------------
__global__ __launch_bounds__(256) void post_kernel(const float* __restrict__ o,
    const float* __restrict__ rbuf, const float* __restrict__ kbuf,
    const float* __restrict__ vbuf, const float* __restrict__ gate,
    const float* __restrict__ r_k, const float* __restrict__ lnx_w,
    const float* __restrict__ lnx_b, bf16_t* __restrict__ obf) {
  __shared__ float red[256];
  int bh = blockIdx.x;
  int m = bh / kH, h = bh % kH, tid = threadIdx.x;
  size_t ov = (size_t)m * kV + h * kDV + tid;
  float xv = o[ov];
  red[tid] = xv;
  __syncthreads();
  for (int s = 128; s > 0; s >>= 1) { if (tid < s) red[tid] += red[tid + s]; __syncthreads(); }
  float mu = red[0] * (1.0f / kDV);
  __syncthreads();
  float d = xv - mu;
  red[tid] = d * d;
  __syncthreads();
  for (int s = 128; s > 0; s >>= 1) { if (tid < s) red[tid] += red[tid + s]; __syncthreads(); }
  float rstd = rsqrtf(red[0] * (1.0f / kDV) + 64e-5f);
  __syncthreads();
  float pb = 0.0f;
  if (tid < kDK) {
    size_t mkidx = (size_t)m * kK + h * kDK + tid;
    pb = rbuf[mkidx] * kbuf[mkidx] * r_k[h * kDK + tid];
  }
  red[tid] = pb;
  __syncthreads();
  for (int s = 128; s > 0; s >>= 1) { if (tid < s) red[tid] += red[tid + s]; __syncthreads(); }
  float bonus = red[0];
  int hv = h * kDV + tid;
  float gn = d * rstd * lnx_w[hv] + lnx_b[hv];
  obf[ov] = (bf16_t)((gn + bonus * vbuf[ov]) * gate[ov]);
}

// ------------------------------- launch -------------------------------
extern "C" void kernel_launch(void* const* d_in, const int* in_sizes, int n_in,
                              void* d_out, int out_size, void* d_ws, size_t ws_size,
                              hipStream_t stream) {
  const float* x    = (const float*)d_in[0];
  const float* ln_w = (const float*)d_in[1];
  const float* ln_b = (const float*)d_in[2];
  const float* x_r  = (const float*)d_in[3];
  const float* x_w  = (const float*)d_in[4];
  const float* x_k  = (const float*)d_in[5];
  const float* x_v  = (const float*)d_in[6];
  const float* x_a  = (const float*)d_in[7];
  const float* x_g  = (const float*)d_in[8];
  const float* W_r  = (const float*)d_in[9];
  const float* W_k  = (const float*)d_in[10];
  const float* W_v  = (const float*)d_in[11];
  const float* W_o  = (const float*)d_in[12];
  const float* w0   = (const float*)d_in[13];
  const float* w1   = (const float*)d_in[14];
  const float* w2   = (const float*)d_in[15];
  const float* a0   = (const float*)d_in[16];
  const float* a1   = (const float*)d_in[17];
  const float* a2   = (const float*)d_in[18];
  const float* g1   = (const float*)d_in[19];
  const float* g2   = (const float*)d_in[20];
  const float* k_k  = (const float*)d_in[21];
  const float* k_a  = (const float*)d_in[22];
  const float* r_k  = (const float*)d_in[23];
  const float* lnxw = (const float*)d_in[24];
  const float* lnxb = (const float*)d_in[25];
  float* out = (float*)d_out;

  char* p = (char*)d_ws;
  auto alloc = [&](size_t bytes) -> void* {
    void* r = (void*)p;
    p += (bytes + 255) & ~(size_t)255;
    return r;
  };
  const size_t MC = (size_t)kM * kC;
  const size_t MK = (size_t)kM * kK;
  const size_t MV = (size_t)kM * kV;

  float*  n_buf = (float*)alloc(MC * 4);
  bf16_t* xr = (bf16_t*)alloc(MC * 2);
  bf16_t* xw = (bf16_t*)alloc(MC * 2);
  bf16_t* xk = (bf16_t*)alloc(MC * 2);
  bf16_t* xv = (bf16_t*)alloc(MC * 2);
  bf16_t* xa = (bf16_t*)alloc(MC * 2);
  bf16_t* xg = (bf16_t*)alloc(MC * 2);
  bf16_t* Wt_r  = (bf16_t*)alloc((size_t)kC * kK * 2);
  bf16_t* Wt_k  = (bf16_t*)alloc((size_t)kC * kK * 2);
  bf16_t* Wt_v  = (bf16_t*)alloc((size_t)kC * kV * 2);
  bf16_t* Wt_o  = (bf16_t*)alloc((size_t)kV * kC * 2);
  bf16_t* Wt_w1 = (bf16_t*)alloc((size_t)kC * 64 * 2);
  bf16_t* Wt_w2 = (bf16_t*)alloc((size_t)64 * kK * 2);
  bf16_t* Wt_a1 = (bf16_t*)alloc((size_t)kC * 64 * 2);
  bf16_t* Wt_a2 = (bf16_t*)alloc((size_t)64 * kK * 2);
  bf16_t* Wt_g1 = (bf16_t*)alloc((size_t)kC * 128 * 2);
  bf16_t* Wt_g2 = (bf16_t*)alloc((size_t)128 * kV * 2);
  float* rbuf   = (float*)alloc(MK * 4);
  float* kbuf   = (float*)alloc(MK * 4);
  float* kkbuf  = (float*)alloc(MK * 4);
  float* dbuf   = (float*)alloc(MK * 4);   // wlog then decay (in place)
  float* abuf   = (float*)alloc(MK * 4);   // alog then sigmoid (in place)
  float* vbuf   = (float*)alloc(MV * 4);
  float* gbuf   = (float*)alloc(MV * 4);   // gate
  float* obuf_o = (float*)alloc(MV * 4);   // scan output
  float* tw     = (float*)alloc((size_t)kM * 64 * 4);
  bf16_t* twb   = (bf16_t*)alloc((size_t)kM * 64 * 2);
  float* ta     = (float*)alloc((size_t)kM * 64 * 4);
  bf16_t* tab   = (bf16_t*)alloc((size_t)kM * 64 * 2);
  float* tg     = (float*)alloc((size_t)kM * 128 * 4);
  bf16_t* tgb   = (bf16_t*)alloc((size_t)kM * 128 * 2);
  bf16_t* obf   = (bf16_t*)alloc(MV * 2);

  auto tconv = [&](const float* src, bf16_t* dst, int R, int Cc2) {
    int cnt = R * Cc2;
    tconv_kernel<<<(cnt + 255) / 256, 256, 0, stream>>>(src, dst, R, Cc2);
  };
  tconv(W_r, Wt_r, kC, kK);
  tconv(W_k, Wt_k, kC, kK);
  tconv(W_v, Wt_v, kC, kV);
  tconv(W_o, Wt_o, kV, kC);
  tconv(w1, Wt_w1, kC, 64);
  tconv(w2, Wt_w2, 64, kK);
  tconv(a1, Wt_a1, kC, 64);
  tconv(a2, Wt_a2, 64, kK);
  tconv(g1, Wt_g1, kC, 128);
  tconv(g2, Wt_g2, 128, kV);

  ln_kernel<<<kM, 256, 0, stream>>>(x, ln_w, ln_b, n_buf);
  mix_kernel<<<kM, 256, 0, stream>>>(n_buf, x_r, x_w, x_k, x_v, x_a, x_g,
                                     xr, xw, xk, xv, xa, xg);

  auto gemm = [&](const bf16_t* A, const bf16_t* Bt, float* Cout,
                  const float* resid, int N, int Kd) {
    if ((N & 127) == 0) {
      dim3 g(N / 128, kM / 128);
      gemm_bf16_kernel<128, 128><<<g, 256, 0, stream>>>(A, Bt, Cout, resid, N, Kd);
    } else {
      dim3 g(N / 64, kM / 64);
      gemm_bf16_kernel<64, 64><<<g, 256, 0, stream>>>(A, Bt, Cout, resid, N, Kd);
    }
  };
  gemm(xr, Wt_r, rbuf, nullptr, kK, kC);
  gemm(xk, Wt_k, kbuf, nullptr, kK, kC);
  gemm(xv, Wt_v, vbuf, nullptr, kV, kC);
  gemm(xw, Wt_w1, tw, nullptr, 64, kC);
  gemm(xa, Wt_a1, ta, nullptr, 64, kC);
  gemm(xg, Wt_g1, tg, nullptr, 128, kC);

  int c64 = kM * 64, c128 = kM * 128;
  act_bf16_kernel<<<(c64 + 255) / 256, 256, 0, stream>>>(tw, twb, c64, 0);   // tanh
  gemm(twb, Wt_w2, dbuf, nullptr, kK, 64);
  decay_kernel<<<((int)MK + 255) / 256, 256, 0, stream>>>(dbuf, w0, dbuf, (int)MK);

  act_bf16_kernel<<<(c64 + 255) / 256, 256, 0, stream>>>(ta, tab, c64, 2);   // copy
  gemm(tab, Wt_a2, abuf, nullptr, kK, 64);
  asig_kernel<<<((int)MK + 255) / 256, 256, 0, stream>>>(abuf, a0, abuf, (int)MK);

  act_bf16_kernel<<<(c128 + 255) / 256, 256, 0, stream>>>(tg, tgb, c128, 1); // sigmoid
  gemm(tgb, Wt_g2, gbuf, nullptr, kV, 128);

  kprep_kernel<<<kM * kH, 128, 0, stream>>>(kbuf, abuf, kkbuf, k_k, k_a);
  scan_kernel<<<kB * kH, 256, 0, stream>>>(rbuf, kbuf, kkbuf, dbuf, abuf, vbuf, obuf_o);
  post_kernel<<<kM * kH, 256, 0, stream>>>(obuf_o, rbuf, kbuf, vbuf, gbuf,
                                           r_k, lnxw, lnxb, obf);

  gemm(obf, Wt_o, out, x, kC, kV);   // out = x + (o*gate) @ W_o
}
